// PointNet2PartSeg_63711544868904
// MI455X (gfx1250) — matrix operand order repaired and checked by reading, not compiled, not run on hardware
//
#include <hip/hip_runtime.h>
#include <stdint.h>
#include <stddef.h>

// PointNet++ part-seg forward for MI455X (gfx1250, wave32).
// All 1x1-conv/BN/ReLU layers run as WMMA GEMMs (v_wmma_f32_16x16x32_f16,
// f16 operands / f32 accumulate, BN folded to per-channel scale+bias in f32).
// GEMM is register-blocked 4x along Cout: one activation (B) fragment feeds
// four weight (A) fragments -> 4 independent WMMAs per K-step.
// Workspace requirement: ~310 MB (two [16][32768][128] f16 ping-pong buffers).

#define BB   16
#define NN   4096
#define NCLS 50

typedef _Float16 h16;
typedef __attribute__((ext_vector_type(16))) _Float16 v16h;
typedef __attribute__((ext_vector_type(8)))  float    v8f;

union V16H { v16h v; uint32_t u[8]; };

// ---------------------------------------------------------------------------
// WMMA GEMM:  D[cout, pt] = sum_k Wq[cout,k] * X[pt,k], then y = sc*d + cb,
// optional ReLU.
// Block = 256 threads = 8 waves; wave w owns M-tile (blockIdx.x*8+w)*16.
// Each wave computes a 64(cout) x 16(pt) strip as 4 accumulators, reusing the
// activation fragment across the 4 weight fragments.
// A-frag (16x32 f16, ISA 7.12.2): lane&15 = row(cout), K = hi*8+(v&3)*2+(v>>2)*16
// B-frag (32x16 f16):             lane&15 = col(pt),   K = hi*8+(v&3)*2+(v>>2)*16
// D      (16x16 f32):             m = i + hi*8,        n = lane&15
// ---------------------------------------------------------------------------
__global__ __launch_bounds__(256) void gemm_bn_act(
    const h16* __restrict__ X,     // [B][M][Kpad] point-major activations
    const h16* __restrict__ Wq,    // [CoutPad][Kpad] f16 weights (zero padded)
    const float* __restrict__ sc, const float* __restrict__ cb,
    h16* __restrict__ Yh,          // [B][M][Cstride]   (used if non-null)
    float* __restrict__ Yf,        // [B][M][CoutStore] (used if Yh==null)
    int M, int Kpad, int CoutStore, int Cstride, int relu)
{
  const int lane  = threadIdx.x & 31;
  const int wave  = threadIdx.x >> 5;
  const int r     = lane & 15;
  const int hi    = lane >> 4;
  const int mBase = (blockIdx.x * 8 + wave) << 4;   // 16 points per wave
  const int oBase = blockIdx.y << 6;                // 64 couts per wave
  const int b     = blockIdx.z;

  const h16* xrow = X + ((size_t)b * M + (size_t)(mBase + r)) * Kpad;
  const h16* wrow0 = Wq + (size_t)(oBase + r)      * Kpad;
  const h16* wrow1 = Wq + (size_t)(oBase + 16 + r) * Kpad;
  const h16* wrow2 = Wq + (size_t)(oBase + 32 + r) * Kpad;
  const h16* wrow3 = Wq + (size_t)(oBase + 48 + r) * Kpad;

  v8f acc0 = {}, acc1 = {}, acc2 = {}, acc3 = {};
  for (int k0 = 0; k0 < Kpad; k0 += 32) {
    V16H bx, a0, a1, a2, a3;
#pragma unroll
    for (int v = 0; v < 8; ++v) {
      const int kA = k0 + hi * 8  + ((v & 3) << 1) + ((v >> 2) << 4);
      const int kB = k0 + hi * 8  + ((v & 3) << 1) + ((v >> 2) << 4);
      bx.u[v] = *(const uint32_t*)(xrow  + kB);   // 2 consecutive f16 -> u32
      a0.u[v] = *(const uint32_t*)(wrow0 + kA);
      a1.u[v] = *(const uint32_t*)(wrow1 + kA);
      a2.u[v] = *(const uint32_t*)(wrow2 + kA);
      a3.u[v] = *(const uint32_t*)(wrow3 + kA);
    }
    acc0 = __builtin_amdgcn_wmma_f32_16x16x32_f16(false, a0.v, false, bx.v,
                                                  (short)0, acc0, false, false);
    acc1 = __builtin_amdgcn_wmma_f32_16x16x32_f16(false, a1.v, false, bx.v,
                                                  (short)0, acc1, false, false);
    acc2 = __builtin_amdgcn_wmma_f32_16x16x32_f16(false, a2.v, false, bx.v,
                                                  (short)0, acc2, false, false);
    acc3 = __builtin_amdgcn_wmma_f32_16x16x32_f16(false, a3.v, false, bx.v,
                                                  (short)0, acc3, false, false);
  }

  const size_t pt = (size_t)b * M + (size_t)(mBase + r);
  v8f accs[4] = {acc0, acc1, acc2, acc3};
#pragma unroll
  for (int j = 0; j < 4; ++j) {
#pragma unroll
    for (int i = 0; i < 8; ++i) {
      const int o = oBase + j * 16 + i + hi * 8;
      float y = sc[o] * accs[j][i] + cb[o];
      if (relu) y = fmaxf(y, 0.f);
      if (Yh)                 Yh[pt * Cstride   + o] = (h16)y;
      else if (o < CoutStore) Yf[pt * CoutStore + o] = y;
    }
  }
}

// --------------------------- weight / BN folding ---------------------------
__global__ void prep_layer(const float* __restrict__ W, const float* __restrict__ bias,
                           const float* __restrict__ gamma, const float* __restrict__ beta,
                           const float* __restrict__ mean, const float* __restrict__ var,
                           h16* __restrict__ Wq, float* __restrict__ sc, float* __restrict__ cb,
                           int Cout, int K, int Kpad, int CoutPad)
{
  int i = blockIdx.x * blockDim.x + threadIdx.x;
  int total = CoutPad * Kpad;
  if (i < total) {
    int o = i / Kpad, k = i % Kpad;
    float w = (o < Cout && k < K) ? W[o * K + k] : 0.f;
    Wq[i] = (h16)w;
  }
  if (i < CoutPad) {
    if (i < Cout) {
      float s, c;
      if (gamma) { s = gamma[i] * rsqrtf(var[i] + 1e-5f);
                   c = s * (bias[i] - mean[i]) + beta[i]; }
      else       { s = 1.f; c = bias[i]; }
      sc[i] = s; cb[i] = c;
    } else { sc[i] = 1.f; cb[i] = 0.f; }
  }
}

// ------------------------------- glue kernels ------------------------------
__global__ void xyz_transpose(const float* __restrict__ src, float* __restrict__ dst,
                              int N, int total)
{ // [B][3][N] -> [B][N][3]
  int t = blockIdx.x * blockDim.x + threadIdx.x; if (t >= total) return;
  int c = t % 3, n = (t / 3) % N, b = t / (3 * N);
  dst[t] = src[((size_t)b * 3 + c) * N + n];
}

__global__ __launch_bounds__(256) void fps_kernel(const float* __restrict__ xyz, // [B][N][3]
                                                  int N, int np, int* __restrict__ out,
                                                  float* __restrict__ dist)      // [B][N]
{
  const int b = blockIdx.x, tid = threadIdx.x;
  __shared__ float s_val[256]; __shared__ int s_idx[256];
  __shared__ float cpt[3];     __shared__ int s_far;
  const float* px = xyz + (size_t)b * N * 3;
  float* d = dist + (size_t)b * N;
  for (int n = tid; n < N; n += 256) d[n] = 1e10f;
  if (tid == 0) s_far = 0;
  __syncthreads();
  for (int it = 0; it < np; ++it) {
    int far = s_far;
    if (tid == 0) { out[b * np + it] = far;
      cpt[0] = px[far*3]; cpt[1] = px[far*3+1]; cpt[2] = px[far*3+2]; }
    __syncthreads();
    float cx = cpt[0], cy = cpt[1], cz = cpt[2];
    float best = -1.f; int bi = 0;
    for (int n = tid; n < N; n += 256) {
      float dx = px[n*3] - cx, dy = px[n*3+1] - cy, dz = px[n*3+2] - cz;
      float nd = fminf(d[n], dx*dx + dy*dy + dz*dz);
      d[n] = nd;
      if (nd > best) { best = nd; bi = n; }
    }
    s_val[tid] = best; s_idx[tid] = bi;
    __syncthreads();
    for (int off = 128; off > 0; off >>= 1) {
      if (tid < off) {
        float v2 = s_val[tid+off]; int i2 = s_idx[tid+off];
        if (v2 > s_val[tid] || (v2 == s_val[tid] && i2 < s_idx[tid]))
          { s_val[tid] = v2; s_idx[tid] = i2; }
      }
      __syncthreads();
    }
    if (tid == 0) s_far = s_idx[0];
    __syncthreads();
  }
}

__global__ void gather_xyz(const float* __restrict__ xyz, const int* __restrict__ fidx,
                           float* __restrict__ out, int N, int S, int total)
{
  int t = blockIdx.x * blockDim.x + threadIdx.x; if (t >= total) return;
  int c = t % 3, s = (t / 3) % S, b = t / (3 * S);
  out[t] = xyz[((size_t)b * N + fidx[b * S + s]) * 3 + c];
}

// One wave32 per query point: ballot-compacted first-nsample-in-radius,
// padded with the first in-radius index (matches sort-then-take semantics).
__global__ __launch_bounds__(256) void ball_query(const float* __restrict__ xyz,  // [B][N][3]
                                                  const float* __restrict__ qxyz, // [B][S][3]
                                                  int N, int S, int ns, float r2,
                                                  int* __restrict__ out, int nq)
{
  const int warp = threadIdx.x >> 5, lane = threadIdx.x & 31;
  const int q = blockIdx.x * (blockDim.x >> 5) + warp;
  if (q >= nq) return;
  const int b = q / S;
  const float* px = xyz + (size_t)b * N * 3;
  const float* pq = qxyz + (size_t)q * 3;
  const float qx = pq[0], qy = pq[1], qz = pq[2];
  int* o = out + (size_t)q * ns;
  int written = 0, first = -1;
  for (int base = 0; base < N && written < ns; base += 32) {
    int n = base + lane;
    bool in = false;
    if (n < N) {
      float dx = px[n*3] - qx, dy = px[n*3+1] - qy, dz = px[n*3+2] - qz;
      in = (dx*dx + dy*dy + dz*dz) <= r2;
    }
    unsigned mask = (unsigned)__ballot(in);
    if (first < 0 && mask) first = base + __builtin_ctz(mask);
    if (in) {
      int pos = written + __builtin_popcount(mask & ((1u << lane) - 1u));
      if (pos < ns) o[pos] = n;
    }
    written += __builtin_popcount(mask);
  }
  if (first < 0) first = N - 1;
  if (written > ns) written = ns;
  for (int slot = written + lane; slot < ns; slot += 32) o[slot] = first;
}

// grouped features: [rel-xyz(3) | pts(Cp)] zero-padded to Cpad, point-major f16
__global__ void group_kernel(const float* __restrict__ xyz,      // [B][N][3]
                             const float* __restrict__ new_xyz,  // [B][S][3]
                             const h16* __restrict__ pts,        // [B][N][Cp] or null
                             const int* __restrict__ idx,        // [B][S][ns]
                             h16* __restrict__ out,              // [B][S*ns][Cpad]
                             int N, int S, int ns, int Cp, int Cpad, int total)
{
  int t = blockIdx.x * blockDim.x + threadIdx.x; if (t >= total) return;
  int s = (t / ns) % S, b = t / (ns * S);
  int id = idx[t];
  const float* pc = new_xyz + ((size_t)b * S + s) * 3;
  const float* pp = xyz + ((size_t)b * N + id) * 3;
  h16* o = out + (size_t)t * Cpad;
  o[0] = (h16)(pp[0] - pc[0]); o[1] = (h16)(pp[1] - pc[1]); o[2] = (h16)(pp[2] - pc[2]);
  int c = 3;
  if (pts) {
    const h16* f = pts + ((size_t)b * N + id) * Cp;
    for (int k = 0; k < Cp; ++k) o[3 + k] = f[k];
    c = 3 + Cp;
  }
  for (; c < Cpad; ++c) o[c] = (h16)0.f;
}

__global__ void pool_max(const h16* __restrict__ X, h16* __restrict__ Y,
                         int S, int ns, int C, int total)
{ // Y[b][s][c] = max_j X[b][s*ns+j][c]
  int t = blockIdx.x * blockDim.x + threadIdx.x; if (t >= total) return;
  int c = t % C, s = (t / C) % S, b = t / (C * S);
  const h16* x = X + (((size_t)b * S + s) * ns) * C + c;
  float m = (float)x[0];
  for (int j = 1; j < ns; ++j) m = fmaxf(m, (float)x[(size_t)j * C]);
  Y[t] = (h16)m;
}

__global__ void fill_h(h16* __restrict__ p, int n)
{ int i = blockIdx.x * blockDim.x + threadIdx.x; if (i < n) p[i] = (h16)0.f; }

__global__ void copy_xyz_h(const float* __restrict__ xyz, h16* __restrict__ dst,
                           int M, int Cstride, int total)
{ // [B][M][3] f32 -> channels 0..2 of [B][M][Cstride] f16
  int t = blockIdx.x * blockDim.x + threadIdx.x; if (t >= total) return;
  int c = t % 3, m = (t / 3) % M, b = t / (3 * M);
  dst[((size_t)b * M + m) * Cstride + c] = (h16)xyz[t];
}

__global__ void copy_feat(const h16* __restrict__ src, h16* __restrict__ dst,
                          int M, int Cs, int Cstride, int dstOff, int total)
{
  int t = blockIdx.x * blockDim.x + threadIdx.x; if (t >= total) return;
  int c = t % Cs, m = (t / Cs) % M, b = t / (Cs * M);
  dst[((size_t)b * M + m) * Cstride + dstOff + c] = src[t];
}

__global__ void bcast_feat(const h16* __restrict__ p2, h16* __restrict__ dst,
                           int N1, int C, int Cstride, int dstOff, int total)
{ // broadcast [B][1][C] to all N1 points
  int t = blockIdx.x * blockDim.x + threadIdx.x; if (t >= total) return;
  int c = t % C, n = (t / C) % N1, b = t / (C * N1);
  dst[((size_t)b * N1 + n) * Cstride + dstOff + c] = p2[(size_t)b * C + c];
}

__global__ void knn3(const float* __restrict__ x1, const float* __restrict__ x2,
                     int N1, int S, int* __restrict__ oidx, float* __restrict__ ow, int total)
{
  int t = blockIdx.x * blockDim.x + threadIdx.x; if (t >= total) return;
  int n = t % N1, b = t / N1;
  const float* p = x1 + ((size_t)b * N1 + n) * 3;
  const float* q = x2 + (size_t)b * S * 3;
  float px = p[0], py = p[1], pz = p[2];
  float d0 = 1e30f, d1 = 1e30f, d2 = 1e30f; int i0 = 0, i1 = 0, i2 = 0;
  for (int s = 0; s < S; ++s) {
    float dx = q[s*3] - px, dy = q[s*3+1] - py, dz = q[s*3+2] - pz;
    float d = dx*dx + dy*dy + dz*dz;
    if      (d < d0) { d2=d1;i2=i1; d1=d0;i1=i0; d0=d;i0=s; }
    else if (d < d1) { d2=d1;i2=i1; d1=d;i1=s; }
    else if (d < d2) { d2=d;i2=s; }
  }
  float w0 = 1.f/fmaxf(d0,1e-10f), w1 = 1.f/fmaxf(d1,1e-10f), w2 = 1.f/fmaxf(d2,1e-10f);
  float sw = w0 + w1 + w2;
  oidx[t*3+0]=i0; oidx[t*3+1]=i1; oidx[t*3+2]=i2;
  ow[t*3+0]=w0/sw; ow[t*3+1]=w1/sw; ow[t*3+2]=w2/sw;
}

__global__ void interp3(const h16* __restrict__ p2, const int* __restrict__ idx3,
                        const float* __restrict__ w3, h16* __restrict__ dst,
                        int N1, int S, int C, int Cstride, int dstOff, int total)
{
  int t = blockIdx.x * blockDim.x + threadIdx.x; if (t >= total) return;
  int c = t % C, n = (t / C) % N1, b = t / (C * N1);
  size_t kb = ((size_t)b * N1 + n) * 3;
  const h16* base = p2 + (size_t)b * S * C;
  float v = w3[kb+0] * (float)base[(size_t)idx3[kb+0]*C + c]
          + w3[kb+1] * (float)base[(size_t)idx3[kb+1]*C + c]
          + w3[kb+2] * (float)base[(size_t)idx3[kb+2]*C + c];
  dst[((size_t)b * N1 + n) * Cstride + dstOff + c] = (h16)v;
}

__global__ void log_softmax_k(const float* __restrict__ logits, // [B][N][C]
                              float* __restrict__ out,          // [B][C][N]
                              int N, int C, int total)
{
  int t = blockIdx.x * blockDim.x + threadIdx.x; if (t >= total) return;
  int n = t % N, b = t / N;
  const float* x = logits + ((size_t)b * N + n) * C;
  float mx = x[0];
  for (int c = 1; c < C; ++c) mx = fmaxf(mx, x[c]);
  float s = 0.f;
  for (int c = 0; c < C; ++c) s += expf(x[c] - mx);
  float lse = mx + logf(s);
  for (int c = 0; c < C; ++c) out[((size_t)b * C + c) * N + n] = x[c] - lse;
}

// ------------------------------- host side ---------------------------------
static inline int nblk (long long t) { return (int)((t + 255) / 256); }
static inline int nblk8(long long q) { return (int)((q + 7) / 8); }

static inline void launch_gemm(hipStream_t s, const h16* X, const h16* Wq,
                               const float* sc, const float* cb,
                               h16* Yh, float* Yf, int M, int Kpad,
                               int CoutPad, int CoutStore, int Cstride, int relu)
{
  dim3 g(M / 128, CoutPad / 64, BB);   // 8 waves x 16 pts, 64 couts per wave
  gemm_bn_act<<<g, 256, 0, s>>>(X, Wq, sc, cb, Yh, Yf, M, Kpad, CoutStore, Cstride, relu);
}

extern "C" void kernel_launch(void* const* d_in, const int* in_sizes, int n_in,
                              void* d_out, int out_size, void* d_ws, size_t ws_size,
                              hipStream_t stream)
{
  (void)in_sizes; (void)n_in; (void)out_size; (void)ws_size;
  // Inputs in setup_inputs() insertion order: xyz, then for each MLP layer
  // {W,b,gamma,beta,mean,var}: sa1(3) sa2(3) sa3(3) fp3(2) fp2(2) fp1(3)
  // head1(1), finally conv2 {W,b}.
  int ii = 0;
  const float* xyz_in = (const float*)d_in[ii++];
  static const int dims[17][2] = {
    {64,3},{64,64},{128,64},
    {128,131},{128,128},{256,128},
    {256,259},{256,256},{1024,256},
    {256,1280},{256,256},
    {256,384},{128,256},
    {128,128},{128,128},{128,128},
    {128,128}};
  const float *Wl[17], *bl[17], *gl[17], *bel[17], *ml[17], *vl[17];
  for (int l = 0; l < 17; ++l) {
    Wl[l]=(const float*)d_in[ii++]; bl[l]=(const float*)d_in[ii++];
    gl[l]=(const float*)d_in[ii++]; bel[l]=(const float*)d_in[ii++];
    ml[l]=(const float*)d_in[ii++]; vl[l]=(const float*)d_in[ii++];
  }
  const float* W2 = (const float*)d_in[ii++];
  const float* b2 = (const float*)d_in[ii++];

  char* base = (char*)d_ws; size_t off = 0;
  auto alloc = [&](size_t bytes) -> void* {
    void* p = base + off; off = (off + bytes + 255) & ~(size_t)255; return p; };

  h16* wq[18]; float* scp[18]; float* cbp[18]; int kpad[18];
  for (int l = 0; l < 17; ++l) {
    int Co = dims[l][0], K = dims[l][1], Kp = (K + 31) & ~31;
    kpad[l] = Kp;
    wq[l]  = (h16*)  alloc((size_t)Co * Kp * sizeof(h16));
    scp[l] = (float*)alloc((size_t)Co * sizeof(float));
    cbp[l] = (float*)alloc((size_t)Co * sizeof(float));
  }
  kpad[17] = 128;
  wq[17]  = (h16*)  alloc((size_t)64 * 128 * sizeof(h16));
  scp[17] = (float*)alloc(64 * sizeof(float));
  cbp[17] = (float*)alloc(64 * sizeof(float));

  float* xyz_pt = (float*)alloc((size_t)BB*NN*3*sizeof(float));
  float* distws = (float*)alloc((size_t)BB*NN*sizeof(float));
  int*   fps1   = (int*)  alloc((size_t)BB*512*sizeof(int));
  float* nx1    = (float*)alloc((size_t)BB*512*3*sizeof(float));
  int*   idx1   = (int*)  alloc((size_t)BB*512*64*sizeof(int));
  int*   fps2   = (int*)  alloc((size_t)BB*128*sizeof(int));
  float* nx2    = (float*)alloc((size_t)BB*128*3*sizeof(float));
  int*   idx2   = (int*)  alloc((size_t)BB*128*64*sizeof(int));
  h16*   l1p    = (h16*)  alloc((size_t)BB*512*128*sizeof(h16));
  h16*   l2p    = (h16*)  alloc((size_t)BB*128*256*sizeof(h16));
  h16*   l3p    = (h16*)  alloc((size_t)BB*1024*sizeof(h16));
  h16*   l2pn   = (h16*)  alloc((size_t)BB*128*256*sizeof(h16));
  h16*   l1pn   = (h16*)  alloc((size_t)BB*512*128*sizeof(h16));
  int*   ki2    = (int*)  alloc((size_t)BB*512*3*sizeof(int));
  float* kw2    = (float*)alloc((size_t)BB*512*3*sizeof(float));
  int*   ki1    = (int*)  alloc((size_t)BB*NN*3*sizeof(int));
  float* kw1    = (float*)alloc((size_t)BB*NN*3*sizeof(float));
  float* logits = (float*)alloc((size_t)BB*NN*NCLS*sizeof(float));
  h16*   actA   = (h16*)  alloc((size_t)BB*32768*128*sizeof(h16));
  h16*   actB   = (h16*)  alloc((size_t)BB*32768*128*sizeof(h16));

  // weight conversion + BN folding
  for (int l = 0; l < 17; ++l) {
    int Co = dims[l][0], K = dims[l][1], Kp = kpad[l];
    prep_layer<<<nblk((long long)Co*Kp),256,0,stream>>>(Wl[l],bl[l],gl[l],bel[l],ml[l],vl[l],
                                                        wq[l],scp[l],cbp[l],Co,K,Kp,Co);
  }
  prep_layer<<<nblk(64*128),256,0,stream>>>(W2,b2,nullptr,nullptr,nullptr,nullptr,
                                            wq[17],scp[17],cbp[17],NCLS,128,128,64);

  xyz_transpose<<<nblk((long long)BB*NN*3),256,0,stream>>>(xyz_in, xyz_pt, NN, BB*NN*3);

  // ---------------- SA1: npoint=512, r=0.2, ns=64, MLP 3->64->64->128 -------
  fps_kernel<<<BB,256,0,stream>>>(xyz_pt, NN, 512, fps1, distws);
  gather_xyz<<<nblk(BB*512*3),256,0,stream>>>(xyz_pt, fps1, nx1, NN, 512, BB*512*3);
  ball_query<<<nblk8(BB*512),256,0,stream>>>(xyz_pt, nx1, NN, 512, 64, 0.04f, idx1, BB*512);
  group_kernel<<<nblk((long long)BB*512*64),256,0,stream>>>(xyz_pt, nx1, (const h16*)nullptr,
                                                            idx1, actA, NN, 512, 64, 0, 32, BB*512*64);
  launch_gemm(stream, actA, wq[0], scp[0], cbp[0], actB, nullptr, 32768, 32,   64,  64,  64, 1);
  launch_gemm(stream, actB, wq[1], scp[1], cbp[1], actA, nullptr, 32768, 64,   64,  64,  64, 1);
  launch_gemm(stream, actA, wq[2], scp[2], cbp[2], actB, nullptr, 32768, 64,  128, 128, 128, 1);
  pool_max<<<nblk((long long)BB*512*128),256,0,stream>>>(actB, l1p, 512, 64, 128, BB*512*128);

  // ---------------- SA2: npoint=128, r=0.4, ns=64, MLP 131->128->128->256 ---
  fps_kernel<<<BB,256,0,stream>>>(nx1, 512, 128, fps2, distws);
  gather_xyz<<<nblk(BB*128*3),256,0,stream>>>(nx1, fps2, nx2, 512, 128, BB*128*3);
  ball_query<<<nblk8(BB*128),256,0,stream>>>(nx1, nx2, 512, 128, 64, 0.16f, idx2, BB*128);
  group_kernel<<<nblk((long long)BB*128*64),256,0,stream>>>(nx1, nx2, l1p, idx2,
                                                            actA, 512, 128, 64, 128, 160, BB*128*64);
  launch_gemm(stream, actA, wq[3], scp[3], cbp[3], actB, nullptr, 8192, 160, 128, 128, 128, 1);
  launch_gemm(stream, actB, wq[4], scp[4], cbp[4], actA, nullptr, 8192, 128, 128, 128, 128, 1);
  launch_gemm(stream, actA, wq[5], scp[5], cbp[5], actB, nullptr, 8192, 128, 256, 256, 256, 1);
  pool_max<<<nblk((long long)BB*128*256),256,0,stream>>>(actB, l2p, 128, 64, 256, BB*128*256);

  // ---------------- SA3 (group_all): MLP 259->256->256->1024 ---------------
  fill_h<<<nblk((long long)BB*128*288),256,0,stream>>>(actA, BB*128*288);
  copy_xyz_h<<<nblk(BB*128*3),256,0,stream>>>(nx2, actA, 128, 288, BB*128*3);
  copy_feat<<<nblk((long long)BB*128*256),256,0,stream>>>(l2p, actA, 128, 256, 288, 3, BB*128*256);
  launch_gemm(stream, actA, wq[6], scp[6], cbp[6], actB, nullptr, 128, 288,  256,  256,  256, 1);
  launch_gemm(stream, actB, wq[7], scp[7], cbp[7], actA, nullptr, 128, 256,  256,  256,  256, 1);
  launch_gemm(stream, actA, wq[8], scp[8], cbp[8], actB, nullptr, 128, 256, 1024, 1024, 1024, 1);
  pool_max<<<nblk((long long)BB*1024),256,0,stream>>>(actB, l3p, 1, 128, 1024, BB*1024);

  // ---------------- FP3: [l2p(256) | bcast l3p(1024)] -> 256 -> 256 --------
  copy_feat<<<nblk((long long)BB*128*256),256,0,stream>>>(l2p, actA, 128, 256, 1280, 0, BB*128*256);
  bcast_feat<<<nblk((long long)BB*128*1024),256,0,stream>>>(l3p, actA, 128, 1024, 1280, 256, BB*128*1024);
  launch_gemm(stream, actA, wq[9],  scp[9],  cbp[9],  actB, nullptr, 128, 1280, 256, 256, 256, 1);
  launch_gemm(stream, actB, wq[10], scp[10], cbp[10], l2pn, nullptr, 128,  256, 256, 256, 256, 1);

  // ---------------- FP2: [l1p(128) | interp(l2pn,256)] -> 256 -> 128 -------
  knn3<<<nblk(BB*512),256,0,stream>>>(nx1, nx2, 512, 128, ki2, kw2, BB*512);
  copy_feat<<<nblk((long long)BB*512*128),256,0,stream>>>(l1p, actA, 512, 128, 384, 0, BB*512*128);
  interp3<<<nblk((long long)BB*512*256),256,0,stream>>>(l2pn, ki2, kw2, actA, 512, 128, 256, 384, 128, BB*512*256);
  launch_gemm(stream, actA, wq[11], scp[11], cbp[11], actB, nullptr, 512, 384, 256, 256, 256, 1);
  launch_gemm(stream, actB, wq[12], scp[12], cbp[12], l1pn, nullptr, 512, 256, 128, 128, 128, 1);

  // ---------------- FP1: interp(l1pn,128) -> 128 x3, head, classifier ------
  knn3<<<nblk((long long)BB*NN),256,0,stream>>>(xyz_pt, nx1, NN, 512, ki1, kw1, BB*NN);
  interp3<<<nblk((long long)BB*NN*128),256,0,stream>>>(l1pn, ki1, kw1, actA, NN, 512, 128, 128, 0, BB*NN*128);
  launch_gemm(stream, actA, wq[13], scp[13], cbp[13], actB, nullptr, NN, 128, 128, 128, 128, 1);
  launch_gemm(stream, actB, wq[14], scp[14], cbp[14], actA, nullptr, NN, 128, 128, 128, 128, 1);
  launch_gemm(stream, actA, wq[15], scp[15], cbp[15], actB, nullptr, NN, 128, 128, 128, 128, 1);
  launch_gemm(stream, actB, wq[16], scp[16], cbp[16], actA, nullptr, NN, 128, 128, 128, 128, 1); // head1
  launch_gemm(stream, actA, wq[17], scp[17], cbp[17], nullptr, logits, NN, 128, 64, NCLS, 0, 0); // conv2

  log_softmax_k<<<nblk((long long)BB*NN),256,0,stream>>>(logits, (float*)d_out, NN, NCLS, BB*NN);
}
